// CoarseEncoder_64269890617429
// MI455X (gfx1250) — compile-verified, matrix operand order checked by reading, not run on hardware
//
#include <hip/hip_runtime.h>
#include <hip/hip_bf16.h>
#include <math.h>

typedef __attribute__((ext_vector_type(16))) _Float16 v16h;
typedef __attribute__((ext_vector_type(8)))  _Float16 v8h;
typedef __attribute__((ext_vector_type(4)))  _Float16 v4h;
typedef __attribute__((ext_vector_type(8)))  float    v8f;

#define NPTS      262144
#define BSC       128           // scenes
#define PPS       2048          // points per scene
#define CIN       256
#define CMID      256
#define COUT      512
#define KTOT      288           // 256 feature + 3 rel + 29 zero pad (9 * 32)
#define ROWS      64            // rows per block in GEMM kernel
#define CHUNKS    (PPS / ROWS)  // 32 chunks per scene
#define ASTRIDE   296           // 288 + 8 pad halfs (keeps 16B align, spreads banks)

// ---------------- ws layout ----------------
// [0)                 W1t f16  : 256 cols x 288 k        = 147456 B
// [147456)            center   : 128 x 4 f32             =   2048 B
// [149504)            partial  : 128 x 32 x 256 f32      = 4194304 B
#define WS_W1T     0
#define WS_CENTER  147456
#define WS_PARTIAL 149504

// ---------------------------------------------------------------------------
// Kernel 0: W1 [259][256] fp32 row-major -> W1t [256 cols][288 k] f16,
//           k >= 259 zero-padded.
// ---------------------------------------------------------------------------
__global__ void prep_w1t_kernel(const float* __restrict__ W1,
                                _Float16* __restrict__ W1t) {
    int idx = blockIdx.x * blockDim.x + threadIdx.x;   // 0 .. 256*288-1
    if (idx >= CMID * KTOT) return;
    int col = idx / KTOT;
    int k   = idx % KTOT;
    float v = (k < CIN + 3) ? W1[(size_t)k * CMID + col] : 0.0f;
    W1t[(size_t)col * KTOT + k] = (_Float16)v;
}

// ---------------------------------------------------------------------------
// Kernel 1: per-scene centroid of pos.  1 block per scene, 256 threads.
// ---------------------------------------------------------------------------
__global__ void centroid_kernel(const float* __restrict__ pos,
                                float* __restrict__ center) {
    int s = blockIdx.x;
    int t = threadIdx.x;
    float sx = 0.f, sy = 0.f, sz = 0.f;
    for (int r = t; r < PPS; r += 256) {
        const float* p = pos + (size_t)(s * PPS + r) * 3;
        sx += p[0]; sy += p[1]; sz += p[2];
    }
    __shared__ float rx[256], ry[256], rz[256];
    rx[t] = sx; ry[t] = sy; rz[t] = sz;
    __syncthreads();
    for (int off = 128; off > 0; off >>= 1) {
        if (t < off) { rx[t] += rx[t + off]; ry[t] += ry[t + off]; rz[t] += rz[t + off]; }
        __syncthreads();
    }
    if (t == 0) {
        const float inv = 1.0f / (float)PPS;
        center[s * 4 + 0] = rx[0] * inv;
        center[s * 4 + 1] = ry[0] * inv;
        center[s * 4 + 2] = rz[0] * inv;
    }
}

// ---------------------------------------------------------------------------
// Kernel 2: fused  [64 rows x 288] @ [288 x 256] WMMA GEMM  + row-max.
// Block = 256 threads (8 waves).  Block b -> rows [b*64, b*64+64) of the
// global point list (scene s = b/32, chunk = b%32).  Wave w computes col
// tiles 2w, 2w+1 over all 4 row tiles; 9 K-steps of v_wmma_f32_16x16x32_f16.
// Writes per-chunk column max (pre-bias, pre-ReLU) to partial[b*256 + col].
// ---------------------------------------------------------------------------
__global__ __launch_bounds__(256)
void gemm_max_kernel(const float* __restrict__ pos,
                     const float* __restrict__ feature,
                     const _Float16* __restrict__ W1t,
                     const float* __restrict__ center,
                     float* __restrict__ partial) {
    __shared__ __align__(16) _Float16 As[ROWS][ASTRIDE];

    const int b = blockIdx.x;
    const int t = threadIdx.x;
    const int s = b >> 5;          // scene
    const int rowbase = b * ROWS;  // global row of first point in chunk

    // ---- stage A: feature -> f16 (64 x 256) ----
    const float4* f4 = (const float4*)feature;
    for (int idx = t; idx < ROWS * (CIN / 4); idx += 256) {
        int r = idx >> 6;          // row in chunk
        int j = idx & 63;          // float4 index along K
        float4 v = f4[(size_t)(rowbase + r) * (CIN / 4) + j];
        v4h h;
        h[0] = (_Float16)v.x; h[1] = (_Float16)v.y;
        h[2] = (_Float16)v.z; h[3] = (_Float16)v.w;
        *(v4h*)&As[r][j * 4] = h;
    }
    // ---- stage A: rel (3) + zero pad to K=288 (and pad cols) ----
    if (t < ROWS) {
        int r = t;
        const float* pp = pos + (size_t)(rowbase + r) * 3;
        float cx = center[s * 4 + 0];
        float cy = center[s * 4 + 1];
        float cz = center[s * 4 + 2];
        As[r][CIN + 0] = (_Float16)(pp[0] - cx);
        As[r][CIN + 1] = (_Float16)(pp[1] - cy);
        As[r][CIN + 2] = (_Float16)(pp[2] - cz);
        for (int k = CIN + 3; k < ASTRIDE; ++k) As[r][k] = (_Float16)0.0f;
    }
    __syncthreads();

    const int lane = t & 31;
    const int wv   = t >> 5;        // wave 0..7
    const int n    = lane & 15;     // M for A-frag, N for B-frag / C-tile
    const int half = lane >> 4;     // 0: lanes 0-15, 1: lanes 16-31
    const int lo8  = half * 8;      // K sub-offset per documented layouts

    v8f acc[4][2] = {};             // 4 row tiles x 2 col tiles

    for (int kt = 0; kt < KTOT / 32; ++kt) {
        const int kb = kt * 32;

        // B fragments (32x16 f16): lane n = col-in-tile; two 16B loads.
        v16h bfrag[2];
#pragma unroll
        for (int ct = 0; ct < 2; ++ct) {
            int col = (wv * 2 + ct) * 16 + n;
            const _Float16* bp = W1t + (size_t)col * KTOT + kb + lo8;
            v8h blo = *(const v8h*)bp;          // K = kb+lo8 .. +7
            v8h bhi = *(const v8h*)(bp + 16);   // K = kb+16+lo8 .. +7
            bfrag[ct] = __builtin_shufflevector(blo, bhi,
                0,1,2,3,4,5,6,7,8,9,10,11,12,13,14,15);
        }

#pragma unroll
        for (int rt = 0; rt < 4; ++rt) {
            // A fragment (16x32 f16): lane n = row-in-tile; two 16B ds loads.
            const _Float16* ap = &As[rt * 16 + n][kb + lo8];
            v8h alo = *(const v8h*)ap;
            v8h ahi = *(const v8h*)(ap + 16);
            v16h afrag = __builtin_shufflevector(alo, ahi,
                0,1,2,3,4,5,6,7,8,9,10,11,12,13,14,15);

            acc[rt][0] = __builtin_amdgcn_wmma_f32_16x16x32_f16(
                false, afrag, false, bfrag[0], (short)0, acc[rt][0], false, false);
            acc[rt][1] = __builtin_amdgcn_wmma_f32_16x16x32_f16(
                false, afrag, false, bfrag[1], (short)0, acc[rt][1], false, false);
        }
    }

    // ---- column max over the 64 rows of this chunk (pre-bias/ReLU) ----
#pragma unroll
    for (int ct = 0; ct < 2; ++ct) {
        float m = acc[0][ct][0];
#pragma unroll
        for (int rt = 0; rt < 4; ++rt)
#pragma unroll
            for (int i = 0; i < 8; ++i)
                m = fmaxf(m, acc[rt][ct][i]);
        // lanes L and L^16 hold the same column, opposite 8-row halves
        m = fmaxf(m, __shfl_xor(m, 16, 32));
        if (half == 0) {
            int col = (wv * 2 + ct) * 16 + n;
            partial[(size_t)b * CMID + col] = m;
        }
    }
}

// ---------------------------------------------------------------------------
// Kernel 3: per scene: max over 32 chunks, +b1, ReLU -> agg; agg@W2+b2;
// split mu/sigma, softplus, rsample.  1 block per scene, 256 threads.
// ---------------------------------------------------------------------------
__global__ __launch_bounds__(256)
void finish_kernel(const float* __restrict__ partial,
                   const float* __restrict__ b1,
                   const float* __restrict__ W2,
                   const float* __restrict__ b2,
                   const float* __restrict__ noise,
                   float* __restrict__ out) {
    __shared__ float agg[CMID];
    const int s = blockIdx.x;
    const int c = threadIdx.x;

    // chunk-max reduction + bias + ReLU (ReLU/bias commute past row-max)
    const float* p = partial + (size_t)s * CHUNKS * CMID + c;
    float m = p[0];
    for (int ch = 1; ch < CHUNKS; ++ch) m = fmaxf(m, p[ch * CMID]);
    agg[c] = fmaxf(m + b1[c], 0.0f);
    __syncthreads();

    // out[s][c] (mu) and out[s][c+256] (sigma_raw)
    float mu = b2[c];
    float sr = b2[c + CMID];
    for (int k = 0; k < CMID; ++k) {
        float a = agg[k];
        mu += a * W2[(size_t)k * COUT + c];
        sr += a * W2[(size_t)k * COUT + c + CMID];
    }
    // stable softplus
    float sig = fmaxf(sr, 0.0f) + log1pf(expf(-fabsf(sr))) + 1e-4f;
    out[(size_t)s * CMID + c] = mu + sig * noise[(size_t)s * CMID + c];
}

// ---------------------------------------------------------------------------
extern "C" void kernel_launch(void* const* d_in, const int* in_sizes, int n_in,
                              void* d_out, int out_size, void* d_ws, size_t ws_size,
                              hipStream_t stream) {
    const float* pos     = (const float*)d_in[0];  // [N,3]
    const float* feature = (const float*)d_in[1];  // [N,256]
    // d_in[2] = batch (sorted repeat(arange(128), 2048)) -- structure known, unused
    const float* W1      = (const float*)d_in[3];  // [259,256]
    const float* b1      = (const float*)d_in[4];  // [256]
    const float* W2      = (const float*)d_in[5];  // [256,512]
    const float* b2      = (const float*)d_in[6];  // [512]
    const float* noise   = (const float*)d_in[7];  // [128,256]
    float* out = (float*)d_out;                    // [128,256]

    _Float16* W1t    = (_Float16*)((char*)d_ws + WS_W1T);
    float*    center = (float*)((char*)d_ws + WS_CENTER);
    float*    partial= (float*)((char*)d_ws + WS_PARTIAL);

    // 0) W1 -> f16, transposed + K-padded to 288
    prep_w1t_kernel<<<(CMID * KTOT + 255) / 256, 256, 0, stream>>>(W1, W1t);
    // 1) centroids
    centroid_kernel<<<BSC, 256, 0, stream>>>(pos, center);
    // 2) fused WMMA GEMM + ReLU-compatible chunk max
    gemm_max_kernel<<<BSC * CHUNKS, 256, 0, stream>>>(pos, feature, W1t, center, partial);
    // 3) finish: scene max, bias/ReLU, second GEMM, softplus, rsample
    finish_kernel<<<BSC, 256, 0, stream>>>(partial, b1, W2, b2, noise, out);
}
